// AttentionLayer_44487271252736
// MI455X (gfx1250) — matrix operand order repaired and checked by reading, not compiled, not run on hardware
//
#include <hip/hip_runtime.h>

// ---------------------------------------------------------------------------
// Multi-head attention layer for MI455X (gfx1250), wave32 + WMMA bf16.
// B=4, N=2048, D=512, H=8, DK=64. All matmuls via v_wmma_f32_16x16x32_bf16.
// Round 4: fully unroll the GEMM K-loops (16 steps) so the ping-pong buffers
// have no loop-carried register constraints -> zero v_dual_mov copies, and
// the scheduler can software-pipeline across the entire reduction.
// ---------------------------------------------------------------------------

#define DEV __device__ __forceinline__

typedef __bf16 bf16_t;
typedef __attribute__((ext_vector_type(16))) __bf16 v16bf;
typedef __attribute__((ext_vector_type(8)))  __bf16 v8bf;
typedef __attribute__((ext_vector_type(8)))  float  v8f;

static constexpr int kB = 4, kN = 2048, kD = 512, kH = 8, kDK = 64;

DEV v16bf cat8(v8bf lo, v8bf hi) {
  return __builtin_shufflevector(lo, hi, 0,1,2,3,4,5,6,7,8,9,10,11,12,13,14,15);
}

DEV v8f wmma_bf16(v16bf a, v16bf b, v8f c) {
  // 8 args: (neg_a, A, neg_b, B, c_mod, C, reuse_a, reuse_b)
  return __builtin_amdgcn_wmma_f32_16x16x32_bf16(false, a, false, b, (short)0, c,
                                                 false, false);
}

// A fragment (16x32 bf16, MxK) from row-major src, leading dim ld (elements).
// ISA layout: lane L holds row (L&15); K elements {ko..ko+7, ko+16..ko+23},
// ko = (L>>4)*8.
DEV v16bf load_a_bf16(const bf16_t* p, int ld, int lane) {
  const int m  = lane & 15;
  const int ko = (lane >> 4) * 8;
  const bf16_t* r = p + (size_t)m * ld + ko;
  v8bf lo = *(const v8bf*)(r);
  v8bf hi = *(const v8bf*)(r + 16);
  return cat8(lo, hi);
}

// A fragment sourced from fp32 row-major data, converted to bf16.
DEV v16bf load_a_f32(const float* p, int ld, int lane) {
  const int m  = lane & 15;
  const int ko = (lane >> 4) * 8;
  const float* r = p + (size_t)m * ld + ko;
  const float4 a0 = ((const float4*)r)[0];
  const float4 a1 = ((const float4*)r)[1];
  const float4 b0 = ((const float4*)(r + 16))[0];
  const float4 b1 = ((const float4*)(r + 16))[1];
  v16bf f;
  f[0]=(bf16_t)a0.x;  f[1]=(bf16_t)a0.y;  f[2]=(bf16_t)a0.z;  f[3]=(bf16_t)a0.w;
  f[4]=(bf16_t)a1.x;  f[5]=(bf16_t)a1.y;  f[6]=(bf16_t)a1.z;  f[7]=(bf16_t)a1.w;
  f[8]=(bf16_t)b0.x;  f[9]=(bf16_t)b0.y;  f[10]=(bf16_t)b0.z; f[11]=(bf16_t)b0.w;
  f[12]=(bf16_t)b1.x; f[13]=(bf16_t)b1.y; f[14]=(bf16_t)b1.z; f[15]=(bf16_t)b1.w;
  return f;
}

// B fragment (32x16, KxN). Memory holds B^T row-major: row n = output column,
// contiguous along K, leading dim ld. ISA layout: lane L holds column (L&15),
// K elements {ko..ko+7, ko+16..ko+23}, ko = (L>>4)*8 (matches A).
DEV v16bf load_b_bf16(const bf16_t* p, int ld, int lane) {
  const int n  = lane & 15;
  const int ko = (lane >> 4) * 8;
  const bf16_t* r = p + (size_t)n * ld + ko;
  return cat8(*(const v8bf*)(r), *(const v8bf*)(r + 16));
}

// ---------------------------------------------------------------------------
// Kernel 0: convert the four DxD fp32 weights into bf16, transposed
// (Wt[n][k] = W[k][n]) so WMMA B-fragments are contiguous.
// ---------------------------------------------------------------------------
__global__ __launch_bounds__(256) void prep_weights(
    const float* __restrict__ Wq, const float* __restrict__ Wk,
    const float* __restrict__ Wv, const float* __restrict__ Wo,
    bf16_t* __restrict__ Wqt, bf16_t* __restrict__ Wkt,
    bf16_t* __restrict__ Wvt, bf16_t* __restrict__ Wot) {
  const int idx   = blockIdx.x * 256 + threadIdx.x;     // 0 .. D*D-1
  const int which = blockIdx.y;
  const float* W  = which == 0 ? Wq : which == 1 ? Wk : which == 2 ? Wv : Wo;
  bf16_t*      Wt = which == 0 ? Wqt : which == 1 ? Wkt : which == 2 ? Wvt : Wot;
  const int n = idx >> 9;         // output column
  const int k = idx & (kD - 1);   // reduction index
  Wt[idx] = (bf16_t)W[(size_t)k * kD + n];
}

// ---------------------------------------------------------------------------
// Kernel 1: QKV projections.  X(8192x512) @ W(512x512) + b, written as bf16.
// mode 0 -> Q/8 (B,H,N,DK)  [softmax scale folded in],
// mode 1 -> K   (B,H,N,DK),
// mode 2 -> V^T (B,H,DK,N).
// Block = 256 threads (8 waves); block tile = 128(M) x 64(N); wave = 16 x 64.
// K reduction fully unrolled (16 k-steps of 32) -> copy-free pipelining.
// ---------------------------------------------------------------------------
__global__ __launch_bounds__(256) void qkv_proj(
    const float* __restrict__ xq, const float* __restrict__ xk,
    const float* __restrict__ xv,
    const bf16_t* __restrict__ Wqt, const bf16_t* __restrict__ Wkt,
    const bf16_t* __restrict__ Wvt,
    const float* __restrict__ bq, const float* __restrict__ bk,
    const float* __restrict__ bv,
    bf16_t* __restrict__ Qd, bf16_t* __restrict__ Kd,
    bf16_t* __restrict__ Vtd) {
  const int mode = blockIdx.z;
  const float*  src  = mode == 0 ? xq : mode == 1 ? xk : xv;
  const bf16_t* Wt   = mode == 0 ? Wqt : mode == 1 ? Wkt : Wvt;
  const float*  bias = mode == 0 ? bq : mode == 1 ? bk : bv;

  const int lane  = threadIdx.x & 31;
  const int wave  = threadIdx.x >> 5;
  const int mBase = blockIdx.x * 128 + wave * 16;
  const int nBase = blockIdx.y * 64;

  const float*  abase = src + (size_t)mBase * kD;
  const bf16_t* wbase = Wt + (size_t)nBase * kD;

  const v8f vzero = {0.f, 0.f, 0.f, 0.f, 0.f, 0.f, 0.f, 0.f};
  v8f acc[4] = {vzero, vzero, vzero, vzero};

#pragma unroll
  for (int kk = 0; kk < kD; kk += 32) {
    const v16bf a = load_a_f32(abase + kk, kD, lane);
#pragma unroll
    for (int t = 0; t < 4; ++t) {
      const v16bf b = load_b_bf16(wbase + (size_t)t * 16 * kD + kk, kD, lane);
      acc[t] = wmma_bf16(a, b, acc[t]);
    }
  }

  const int col     = lane & 15;
  const int rowHalf = (lane >> 4) * 8;
#pragma unroll
  for (int t = 0; t < 4; ++t) {
    const int   n  = nBase + t * 16 + col;
    const float bn = bias[n];
    const int   h  = n >> 6;        // head
    const int   dk = n & (kDK - 1); // dim within head
#pragma unroll
    for (int r = 0; r < 8; ++r) {
      const int m  = mBase + rowHalf + r;
      const int bi = m >> 11;            // batch
      const int s  = m & (kN - 1);       // sequence position
      float v = acc[t][r] + bn;
      if (mode == 0) v *= 0.125f;        // fold 1/sqrt(DK) into Q
      if (mode < 2) {
        bf16_t* dst = (mode == 0) ? Qd : Kd;
        dst[(((size_t)(bi * kH + h)) * kN + s) * kDK + dk] = (bf16_t)v;
      } else {
        Vtd[(((size_t)(bi * kH + h)) * kDK + dk) * kN + s] = (bf16_t)v;
      }
    }
  }
}

// ---------------------------------------------------------------------------
// Online-softmax update for one 16x32 score tile pair (C-layout), including
// the P -> LDS staging for the A-fragment reload.  Inlined.
// ---------------------------------------------------------------------------
DEV void softmax_update(const v8f& s0, const v8f& s1, v8f (&c)[4],
                        float (&mrun)[8], float (&lrun)[8],
                        bf16_t (*lds)[32], int lane) {
  const int col   = lane & 15;
  const int half8 = (lane >> 4) * 8;
#pragma unroll
  for (int r = 0; r < 8; ++r) {
    const float t0 = s0[r];
    const float t1 = s1[r];
    float rm = fmaxf(t0, t1);
    rm = fmaxf(rm, __shfl_xor(rm, 1, 32));
    rm = fmaxf(rm, __shfl_xor(rm, 2, 32));
    rm = fmaxf(rm, __shfl_xor(rm, 4, 32));
    rm = fmaxf(rm, __shfl_xor(rm, 8, 32));
    const float mn = fmaxf(mrun[r], rm);
    const float al = __expf(mrun[r] - mn);
    const float p0 = __expf(t0 - mn);
    const float p1 = __expf(t1 - mn);
    float rs = p0 + p1;
    rs += __shfl_xor(rs, 1, 32);
    rs += __shfl_xor(rs, 2, 32);
    rs += __shfl_xor(rs, 4, 32);
    rs += __shfl_xor(rs, 8, 32);
    lrun[r] = lrun[r] * al + rs;
    mrun[r] = mn;
    c[0][r] *= al; c[1][r] *= al; c[2][r] *= al; c[3][r] *= al;
    const int row = half8 + r;
    lds[row][col]      = (bf16_t)p0;
    lds[row][col + 16] = (bf16_t)p1;
  }
}

// ---------------------------------------------------------------------------
// Kernel 2: flash attention per (b,h).  Block = 8 waves, each wave owns 16
// query rows and a 16x64 f32 accumulator.  KV processed in blocks of 32,
// loop unrolled x2 with ping-pong K-fragment buffers (no register copies).
// ---------------------------------------------------------------------------
__global__ __launch_bounds__(256) void attn_kernel(
    const bf16_t* __restrict__ Qd, const bf16_t* __restrict__ Kd,
    const bf16_t* __restrict__ Vtd, bf16_t* __restrict__ Ctx) {
  __shared__ __align__(64) bf16_t lds_p[8][16][32];  // per-wave P tile

  const int lane  = threadIdx.x & 31;
  const int wave  = threadIdx.x >> 5;
  const int bh    = blockIdx.y;                 // 0..31 = b*H + h
  const int qBase = blockIdx.x * 128 + wave * 16;

  const bf16_t* Qp = Qd  + (size_t)bh * kN * kDK;
  const bf16_t* Kp = Kd  + (size_t)bh * kN * kDK;
  const bf16_t* Vp = Vtd + (size_t)bh * kDK * kN;
  bf16_t (*ldsw)[32] = lds_p[wave];

  // Q fragments for this wave's 16 rows (DK=64 -> two K-steps of 32).
  // Q was pre-scaled by 1/sqrt(DK) in the projection.
  const v16bf qa0 = load_a_bf16(Qp + (size_t)qBase * kDK +  0, kDK, lane);
  const v16bf qa1 = load_a_bf16(Qp + (size_t)qBase * kDK + 32, kDK, lane);

  const v8f vzero = {0.f, 0.f, 0.f, 0.f, 0.f, 0.f, 0.f, 0.f};
  v8f c[4] = {vzero, vzero, vzero, vzero};

  float mrun[8], lrun[8];
#pragma unroll
  for (int r = 0; r < 8; ++r) { mrun[r] = -1e30f; lrun[r] = 0.f; }

  // prologue: K fragments for kv = 0 into buffer A
  v16bf ka0, ka1, ka2, ka3;
  ka0 = load_b_bf16(Kp +  0 * kDK +  0, kDK, lane);
  ka1 = load_b_bf16(Kp +  0 * kDK + 32, kDK, lane);
  ka2 = load_b_bf16(Kp + 16 * kDK +  0, kDK, lane);
  ka3 = load_b_bf16(Kp + 16 * kDK + 32, kDK, lane);

  for (int kv = 0; kv < kN; kv += 64) {
    // =================== sub-block 0: keys kv .. kv+31 =====================
    v8f s0 = wmma_bf16(qa0, ka0, vzero);
    s0     = wmma_bf16(qa1, ka1, s0);
    v8f s1 = wmma_bf16(qa0, ka2, vzero);
    s1     = wmma_bf16(qa1, ka3, s1);

    // refill buffer B with keys kv+32 (always in range) and load this
    // sub-block's V fragments; softmax below hides their latency.
    v16bf kb0 = load_b_bf16(Kp + (size_t)(kv + 32) * kDK +  0, kDK, lane);
    v16bf kb1 = load_b_bf16(Kp + (size_t)(kv + 32) * kDK + 32, kDK, lane);
    v16bf kb2 = load_b_bf16(Kp + (size_t)(kv + 48) * kDK +  0, kDK, lane);
    v16bf kb3 = load_b_bf16(Kp + (size_t)(kv + 48) * kDK + 32, kDK, lane);
    v16bf va0 = load_b_bf16(Vp + (size_t) 0 * kN + kv, kN, lane);
    v16bf va1 = load_b_bf16(Vp + (size_t)16 * kN + kv, kN, lane);
    v16bf va2 = load_b_bf16(Vp + (size_t)32 * kN + kv, kN, lane);
    v16bf va3 = load_b_bf16(Vp + (size_t)48 * kN + kv, kN, lane);

    if (kv + 128 < kN) {  // pull a later block toward the WGP
      __builtin_prefetch(Kp + (size_t)(kv + 128) * kDK + lane * 16, 0, 1);
      __builtin_prefetch(Vp + (size_t)(lane * 2) * kN + kv + 128, 0, 1);
    }

    softmax_update(s0, s1, c, mrun, lrun, ldsw, lane);

    // wave-synchronous LDS RAW: DS ops are in-order per wave; the explicit
    // split-counter wait also stops the compiler hoisting the reload.
    asm volatile("s_wait_dscnt 0" ::: "memory");
    {
      const v16bf pa = load_a_bf16(&ldsw[0][0], 32, lane);
      c[0] = wmma_bf16(pa, va0, c[0]);
      c[1] = wmma_bf16(pa, va1, c[1]);
      c[2] = wmma_bf16(pa, va2, c[2]);
      c[3] = wmma_bf16(pa, va3, c[3]);
    }

    // =================== sub-block 1: keys kv+32 .. kv+63 ==================
    s0 = wmma_bf16(qa0, kb0, vzero);
    s0 = wmma_bf16(qa1, kb1, s0);
    s1 = wmma_bf16(qa0, kb2, vzero);
    s1 = wmma_bf16(qa1, kb3, s1);

    // refill buffer A with keys kv+64 (branch-free tail) + V for kv+32.
    const int kvn = (kv + 64 < kN) ? kv + 64 : kv;
    ka0 = load_b_bf16(Kp + (size_t)(kvn)      * kDK +  0, kDK, lane);
    ka1 = load_b_bf16(Kp + (size_t)(kvn)      * kDK + 32, kDK, lane);
    ka2 = load_b_bf16(Kp + (size_t)(kvn + 16) * kDK +  0, kDK, lane);
    ka3 = load_b_bf16(Kp + (size_t)(kvn + 16) * kDK + 32, kDK, lane);
    v16bf vb0 = load_b_bf16(Vp + (size_t) 0 * kN + kv + 32, kN, lane);
    v16bf vb1 = load_b_bf16(Vp + (size_t)16 * kN + kv + 32, kN, lane);
    v16bf vb2 = load_b_bf16(Vp + (size_t)32 * kN + kv + 32, kN, lane);
    v16bf vb3 = load_b_bf16(Vp + (size_t)48 * kN + kv + 32, kN, lane);

    softmax_update(s0, s1, c, mrun, lrun, ldsw, lane);

    asm volatile("s_wait_dscnt 0" ::: "memory");
    {
      const v16bf pa = load_a_bf16(&ldsw[0][0], 32, lane);
      c[0] = wmma_bf16(pa, vb0, c[0]);
      c[1] = wmma_bf16(pa, vb1, c[1]);
      c[2] = wmma_bf16(pa, vb2, c[2]);
      c[3] = wmma_bf16(pa, vb3, c[3]);
    }
  }

  // epilogue: normalize (reciprocal-multiply) and write ctx as (B,N,H*DK) bf16
  float inv[8];
#pragma unroll
  for (int r = 0; r < 8; ++r) inv[r] = 1.0f / lrun[r];

  const int col   = lane & 15;
  const int half8 = (lane >> 4) * 8;
  const int bi = bh >> 3;
  const int h  = bh & 7;
#pragma unroll
  for (int t = 0; t < 4; ++t) {
#pragma unroll
    for (int r = 0; r < 8; ++r) {
      const int   m = qBase + half8 + r;
      const float v = c[t][r] * inv[r];
      Ctx[((size_t)(bi * kN + m)) * kD + h * kDK + t * 16 + col] = (bf16_t)v;
    }
  }
}

// ---------------------------------------------------------------------------
// Kernel 3: output projection.  ctx(8192x512 bf16) @ Wo + bo -> fp32 out.
// K reduction fully unrolled (16 k-steps of 32).
// ---------------------------------------------------------------------------
__global__ __launch_bounds__(256) void out_proj(
    const bf16_t* __restrict__ Ctx, const bf16_t* __restrict__ Wot,
    const float* __restrict__ bo, float* __restrict__ out) {
  const int lane  = threadIdx.x & 31;
  const int wave  = threadIdx.x >> 5;
  const int mBase = blockIdx.x * 128 + wave * 16;
  const int nBase = blockIdx.y * 64;

  const bf16_t* abase = Ctx + (size_t)mBase * kD;
  const bf16_t* wbase = Wot + (size_t)nBase * kD;

  const v8f vzero = {0.f, 0.f, 0.f, 0.f, 0.f, 0.f, 0.f, 0.f};
  v8f acc[4] = {vzero, vzero, vzero, vzero};

#pragma unroll
  for (int kk = 0; kk < kD; kk += 32) {
    const v16bf a = load_a_bf16(abase + kk, kD, lane);
#pragma unroll
    for (int t = 0; t < 4; ++t) {
      const v16bf b = load_b_bf16(wbase + (size_t)t * 16 * kD + kk, kD, lane);
      acc[t] = wmma_bf16(a, b, acc[t]);
    }
  }

  const int col     = lane & 15;
  const int rowHalf = (lane >> 4) * 8;
#pragma unroll
  for (int t = 0; t < 4; ++t) {
    const int   n  = nBase + t * 16 + col;
    const float bn = bo[n];
#pragma unroll
    for (int r = 0; r < 8; ++r) {
      const int m = mBase + rowHalf + r;
      out[(size_t)m * kD + n] = acc[t][r] + bn;
    }
  }
}

// ---------------------------------------------------------------------------
extern "C" void kernel_launch(void* const* d_in, const int* in_sizes, int n_in,
                              void* d_out, int out_size, void* d_ws,
                              size_t ws_size, hipStream_t stream) {
  (void)in_sizes; (void)n_in; (void)out_size; (void)ws_size;

  const float* query = (const float*)d_in[0];
  const float* key   = (const float*)d_in[1];
  const float* value = (const float*)d_in[2];
  const float* Wq    = (const float*)d_in[3];
  const float* bq    = (const float*)d_in[4];
  const float* Wk    = (const float*)d_in[5];
  const float* bk    = (const float*)d_in[6];
  const float* Wv    = (const float*)d_in[7];
  const float* bv    = (const float*)d_in[8];
  const float* Wo    = (const float*)d_in[9];
  const float* bo    = (const float*)d_in[10];
  float*       out   = (float*)d_out;

  // Workspace carve-up (~34 MB total, every byte rewritten each call).
  char* ws = (char*)d_ws;
  const size_t wBytes = (size_t)kD * kD * sizeof(bf16_t);            // 0.5 MB
  const size_t tBytes = (size_t)kB * kH * kN * kDK * sizeof(bf16_t); // 8 MB
  bf16_t* Wqt = (bf16_t*)ws;  ws += wBytes;
  bf16_t* Wkt = (bf16_t*)ws;  ws += wBytes;
  bf16_t* Wvt = (bf16_t*)ws;  ws += wBytes;
  bf16_t* Wot = (bf16_t*)ws;  ws += wBytes;
  bf16_t* Qd  = (bf16_t*)ws;  ws += tBytes;
  bf16_t* Kd  = (bf16_t*)ws;  ws += tBytes;
  bf16_t* Vtd = (bf16_t*)ws;  ws += tBytes;
  bf16_t* Ctx = (bf16_t*)ws;  ws += tBytes;

  // 0) bf16 transposed weights
  prep_weights<<<dim3(kD * kD / 256, 4), 256, 0, stream>>>(
      Wq, Wk, Wv, Wo, Wqt, Wkt, Wvt, Wot);

  // 1) Q/K/V projections (grid.z selects which)
  qkv_proj<<<dim3(kB * kN / 128, kD / 64, 3), 256, 0, stream>>>(
      query, key, value, Wqt, Wkt, Wvt, bq, bk, bv, Qd, Kd, Vtd);

  // 2) flash attention over 32 (b,h) pairs
  attn_kernel<<<dim3(kN / 128, kB * kH), 256, 0, stream>>>(Qd, Kd, Vtd, Ctx);

  // 3) output projection
  out_proj<<<dim3(kB * kN / 128, kD / 64), 256, 0, stream>>>(Ctx, Wot, bo, out);
}
